// model_996432412787
// MI455X (gfx1250) — matrix operand order repaired and checked by reading, not compiled, not run on hardware
//
#include <hip/hip_runtime.h>
#include <hip/hip_bf16.h>
#include <cstdint>

typedef __attribute__((ext_vector_type(16))) __bf16 v16bf;
typedef __attribute__((ext_vector_type(8)))  float  v8f;

static __device__ __forceinline__ uint16_t f2bf(float f) {
  uint32_t u = __float_as_uint(f);
  uint32_t r = (u + 0x7FFFu + ((u >> 16) & 1u)) >> 16;
  return (uint16_t)r;
}
static __device__ __forceinline__ float bf2f(uint16_t h) {
  return __uint_as_float((uint32_t)h << 16);
}

// ============ input prep: f32 NCHW (10,3,64,64) -> bf16 NHWC (10,64,64,8) ============
__global__ void prep_x_kernel(const float* __restrict__ x, uint16_t* __restrict__ y, int Bn) {
  int idx = blockIdx.x * blockDim.x + threadIdx.x;
  int total = Bn * 64 * 64 * 8;
  if (idx >= total) return;
  int c = idx & 7;
  int t = idx >> 3;
  int w = t & 63;
  t >>= 6;
  int h = t & 63;
  int b = t >> 6;
  uint16_t o = 0;
  if (c < 3) o = f2bf(x[(((size_t)b * 3 + c) * 64 + h) * 64 + w]);
  y[idx] = o;
}

__global__ void zero_pad_kernel(uint16_t* __restrict__ z) {
  z[threadIdx.x] = 0;   // 64 threads, 128-byte zero page
}

// ============ weight packing -> bf16 [Npad][Kpad], K ordered (kh, kw, ic) ============
// mode 0: conv OIHW              src = w[n][ic][kh][kw]            (ICp may pad ICact)
// mode 1: deconv IOHW + flip     src = w[ic][n][KH-1-kh][KW-1-kw]
// mode 2: FC [K][N] transpose    src = w[k][n]
// mode 3: fc1 (K permute)        k=(pos,c): src = w[c*KH + pos][n]         (KH=HW)
// mode 4: fc2 (N permute)        n=(pos,c): src = w[k][c*KH + pos]         (KH=HW)
struct PackP { int mode, Kpad, Kact, Nact, Npad, ICp, ICact, KH, KW; };

__global__ void pack_w_kernel(const float* __restrict__ w, uint16_t* __restrict__ wt, PackP p) {
  int idx = blockIdx.x * blockDim.x + threadIdx.x;
  int total = p.Npad * p.Kpad;
  if (idx >= total) return;
  int n = idx / p.Kpad;
  int k = idx - n * p.Kpad;
  uint16_t o = 0;
  if (n < p.Nact && k < p.Kact) {
    int src = -1;
    if (p.mode == 0 || p.mode == 1) {
      int kwic = p.KW * p.ICp;
      int kh = k / kwic;
      int r  = k - kh * kwic;
      int kw = r / p.ICp;
      int ic = r - kw * p.ICp;
      if (ic < p.ICact) {
        if (p.mode == 0) src = ((n * p.ICact + ic) * p.KH + kh) * p.KW + kw;
        else             src = ((ic * p.Nact + n) * p.KH + (p.KH - 1 - kh)) * p.KW + (p.KW - 1 - kw);
      }
    } else if (p.mode == 2) {
      src = k * p.Nact + n;
    } else if (p.mode == 3) {
      int pos = k / p.ICp, c = k - pos * p.ICp;
      src = (c * p.KH + pos) * p.Nact + n;
    } else { // mode 4
      int pos = n / p.ICp, c = n - pos * p.ICp;
      src = k * p.Nact + (c * p.KH + pos);
    }
    if (src >= 0) o = f2bf(w[src]);
  }
  wt[idx] = o;
}

__global__ void pack_bias_fc2_kernel(const float* __restrict__ b, float* __restrict__ o) {
  int n = blockIdx.x * blockDim.x + threadIdx.x;
  if (n >= 4096) return;
  int pos = n >> 8, c = n & 255;
  o[n] = b[c * 16 + pos];
}

// ============ implicit-GEMM conv/deconv/FC over bf16 NHWC via WMMA ============
// Wave computes a 16(M) x 64(N) tile: 4 accumulators, A reused across 4 WMMAs/step.
struct GemmP {
  unsigned long long icMul;   // magic for k / IC  (shift 40)
  unsigned kwMul;             // magic for s / KW  (shift 8)
  int IC, IH, IW;
  int OC, OH, OW;
  int KW, stride, pad;
  int Kpad, Kact, Mact;
  int relu;
  int outf32;                 // 0: bf16 NHWC, 1: f32 channel-major (B,OC,OH*OW)
};

union Frag { uint4 u[2]; v16bf v; };

template <int DECONV>
__global__ void __launch_bounds__(32) gemm_wmma_kernel(
    const uint16_t* __restrict__ x, const uint16_t* __restrict__ wt,
    const float* __restrict__ bias, void* __restrict__ yout,
    const uint16_t* __restrict__ zpad, GemmP p) {
  const int lane = threadIdx.x;
  const int mt = blockIdx.x * 16;
  const int nt = blockIdx.y * 64;
  const int l16 = lane & 15;
  const int hi  = lane >> 4;

  const int mrow = mt + l16;
  const int ohw  = p.OH * p.OW;

  int b = 0, oh = 0, ow = 0;
  const bool mvalid = (mrow < p.Mact);
  if (mvalid) {
    b = mrow / ohw;
    int r = mrow - b * ohw;
    oh = r / p.OW;
    ow = r - oh * p.OW;
  }
  // hoisted per-lane gather bases
  const int ihb = DECONV ? (oh - 2) : (oh * p.stride - p.pad);
  const int iwb = DECONV ? (ow - 2) : (ow * p.stride - p.pad);
  const uint16_t* xb = x + (size_t)b * p.IH * p.IW * p.IC;
  const long pbase = (long)ihb * p.IW + iwb;   // conv mode: linearized spatial base

  const uint16_t* brow[4];
#pragma unroll
  for (int t = 0; t < 4; ++t)
    brow[t] = wt + (size_t)(nt + t * 16 + l16) * p.Kpad;

  // branchless A-run gather: invalid runs load from the zero page
  auto gather_a = [&](int k) -> uint4 {
    unsigned s = (unsigned)(((unsigned long long)(unsigned)k * p.icMul) >> 40);
    int ic = k - (int)s * p.IC;
    int kh = (int)((s * p.kwMul) >> 8);
    int kw = (int)s - kh * p.KW;
    bool ok;
    long off;
    if (!DECONV) {
      int ih = ihb + kh;
      int iw = iwb + kw;
      ok = ((unsigned)ih < (unsigned)p.IH) & ((unsigned)iw < (unsigned)p.IW);
      off = (pbase + (long)kh * p.IW + kw) * p.IC + ic;
    } else {
      int u = ihb + kh;
      int v = iwb + kw;
      ok = ((u | v) >= 0) & (u < 2 * p.IH) & (v < 2 * p.IW) & (((u | v) & 1) == 0);
      off = ((long)(u >> 1) * p.IW + (v >> 1)) * p.IC + ic;
    }
    ok &= mvalid & (k < p.Kact);
    const uint16_t* ptr = ok ? (xb + off) : zpad;
    return *(const uint4*)ptr;
  };

  auto load_step = [&](int kb, Frag& a, Frag (&bf)[4]) {
    a.u[0] = gather_a(kb + hi * 8);
    a.u[1] = gather_a(kb + hi * 8 + 16);
#pragma unroll
    for (int t = 0; t < 4; ++t) {
      const uint4* bp = (const uint4*)(brow[t] + kb + hi * 8);
      bf[t].u[0] = bp[0];
      bf[t].u[1] = bp[2];
    }
  };

  v8f acc[4] = {{}, {}, {}, {}};
  auto wmma_step = [&](Frag& a, Frag (&bf)[4]) {
#pragma unroll
    for (int t = 0; t < 4; ++t)
      acc[t] = __builtin_amdgcn_wmma_f32_16x16x32_bf16(
          false, a.v, false, bf[t].v, (short)0, acc[t], false, false);
  };

  // double-buffered, software-pipelined K loop (steps >= 2 for all layers)
  const int steps = p.Kpad >> 5;
  Frag aA, aB;
  Frag bA[4], bB[4];
  load_step(0, aA, bA);
  int i = 1;
  for (; i + 1 < steps; i += 2) {
    load_step(i << 5, aB, bB);
    wmma_step(aA, bA);
    load_step((i + 1) << 5, aA, bA);
    wmma_step(aB, bB);
  }
  if (i < steps) {
    load_step(i << 5, aB, bB);
    wmma_step(aA, bA);
    wmma_step(aB, bB);
  } else {
    wmma_step(aA, bA);
  }

  // D layout: VGPR r -> m = mt + r + hi*8, n = l16 within each 16-wide tile
#pragma unroll
  for (int t = 0; t < 4; ++t) {
    const int ncol = nt + t * 16 + l16;
    if (ncol < p.OC) {
      float bv = bias ? bias[ncol] : 0.f;
#pragma unroll
      for (int r = 0; r < 8; ++r) {
        int m = mt + r + hi * 8;
        if (m < p.Mact) {
          float v = acc[t][r] + bv;
          if (p.relu) v = fmaxf(v, 0.f);
          int bb = m / ohw;
          int pos = m - bb * ohw;
          if (p.outf32 == 0) {
            ((uint16_t*)yout)[((size_t)bb * ohw + pos) * p.OC + ncol] = f2bf(v);
          } else {
            ((float*)yout)[((size_t)bb * p.OC + ncol) * ohw + pos] = v;
          }
        }
      }
    }
  }
}

// ============ projective transform + quaternion rotation (reads bf16 NHWC) ============
__global__ void geom_kernel(const uint16_t* __restrict__ xyz, const float* __restrict__ quat,
                            float* __restrict__ out, int Bn) {
  const int V = 8, HW = 128 * 128;
  int idx = blockIdx.x * blockDim.x + threadIdx.x;
  int total = Bn * V * HW;
  if (idx >= total) return;
  int hw = idx % HW;
  int t  = idx / HW;
  int v  = t % V;
  int b  = t / V;

  const uint16_t* px = xyz + ((size_t)b * HW + hw) * 24 + v * 3;
  float X = bf2f(px[0]);
  float Y = bf2f(px[1]);
  float Z = bf2f(px[2]);

  float cx = (X - 32.f) * (1.f / 64.f);
  float cy = (32.f - Y) * (1.f / 64.f);
  float cz = -(Z + 1.0f);

  float w  = quat[v * 4 + 0];
  float qx = quat[v * 4 + 1];
  float qy = quat[v * 4 + 2];
  float qz = quat[v * 4 + 3];
  float R00 = 1.f - 2.f * (qy * qy + qz * qz), R01 = 2.f * (qx * qy - w * qz), R02 = 2.f * (qx * qz + w * qy);
  float R10 = 2.f * (qx * qy + w * qz), R11 = 1.f - 2.f * (qx * qx + qz * qz), R12 = 2.f * (qy * qz - w * qx);
  float R20 = 2.f * (qx * qz - w * qy), R21 = 2.f * (qy * qz + w * qx), R22 = 1.f - 2.f * (qx * qx + qy * qy);

  float o0 = R00 * cx + R10 * cy + R20 * cz;
  float o1 = R01 * cx + R11 * cy + R21 * cz;
  float o2 = R02 * cx + R12 * cy + R22 * cz;

  size_t VHW = (size_t)V * HW;
  size_t o = (size_t)v * HW + hw;
  out[((size_t)b * 3 + 0) * VHW + o] = o0;
  out[((size_t)b * 3 + 1) * VHW + o] = o1;
  out[((size_t)b * 3 + 2) * VHW + o] = o2;
}

// ============ host side ============
static void run_pack(hipStream_t s, const float* w, uint16_t* wt, PackP p) {
  int total = p.Npad * p.Kpad;
  pack_w_kernel<<<(total + 255) / 256, 256, 0, s>>>(w, wt, p);
}

static void run_gemm(hipStream_t s, const uint16_t* x, const uint16_t* wt,
                     const float* bias, void* y, const uint16_t* zpad,
                     GemmP p, int Npad64, int deconv) {
  int Mpad = (p.Mact + 15) & ~15;
  dim3 grid(Mpad / 16, Npad64 / 64), block(32);
  if (deconv)
    gemm_wmma_kernel<1><<<grid, block, 0, s>>>(x, wt, bias, y, zpad, p);
  else
    gemm_wmma_kernel<0><<<grid, block, 0, s>>>(x, wt, bias, y, zpad, p);
}

static GemmP mk_gemm(int IC, int IH, int IW, int OC, int OH, int OW, int KW,
                     int stride, int pad, int Kpad, int Kact, int Mact,
                     int relu, int outf32) {
  GemmP g;
  g.icMul = ((1ull << 40) / (unsigned)IC) + 1;
  g.kwMul = (unsigned)(256 / KW + 1);
  g.IC = IC; g.IH = IH; g.IW = IW; g.OC = OC; g.OH = OH; g.OW = OW;
  g.KW = KW; g.stride = stride; g.pad = pad;
  g.Kpad = Kpad; g.Kact = Kact; g.Mact = Mact;
  g.relu = relu; g.outf32 = outf32;
  return g;
}

static inline int pad64(int n) { return (n + 63) & ~63; }

extern "C" void kernel_launch(void* const* d_in, const int* in_sizes, int n_in,
                              void* d_out, int out_size, void* d_ws, size_t ws_size,
                              hipStream_t stream) {
  (void)in_sizes; (void)n_in; (void)out_size; (void)ws_size;
  const float* x     = (const float*)d_in[0];
  const float* ew[4] = {(const float*)d_in[1], (const float*)d_in[3], (const float*)d_in[5], (const float*)d_in[7]};
  const float* eb[4] = {(const float*)d_in[2], (const float*)d_in[4], (const float*)d_in[6], (const float*)d_in[8]};
  const float* efw   = (const float*)d_in[9];
  const float* efb   = (const float*)d_in[10];
  const float* dfw   = (const float*)d_in[11];
  const float* dfb   = (const float*)d_in[12];
  const float* dw[5] = {(const float*)d_in[13], (const float*)d_in[15], (const float*)d_in[17], (const float*)d_in[19], (const float*)d_in[21]};
  const float* db[5] = {(const float*)d_in[14], (const float*)d_in[16], (const float*)d_in[18], (const float*)d_in[20], (const float*)d_in[22]};
  const float* xyzw  = (const float*)d_in[23];
  const float* xyzb  = (const float*)d_in[24];
  const float* maskw = (const float*)d_in[25];
  const float* maskb = (const float*)d_in[26];
  const float* quat  = (const float*)d_in[27];
  float* out = (float*)d_out;

  const int Bn = 10;
  const size_t BUFSZ = 8u * 1024u * 1024u;  // bf16 elements per ping-pong buffer
  uint16_t* bufs[2];
  bufs[0] = (uint16_t*)d_ws;
  bufs[1] = bufs[0] + BUFSZ;
  float* bias_fc2 = (float*)(bufs[1] + BUFSZ);
  uint16_t* zpad  = (uint16_t*)(bias_fc2 + 4096);   // 64 ushorts, zeroed each launch
  uint16_t* wbase = zpad + 64;
  size_t woff = 0;
  auto walloc = [&](size_t n) { uint16_t* p = wbase + woff; woff += (n + 31) & ~(size_t)31; return p; };

  zero_pad_kernel<<<1, 64, 0, stream>>>(zpad);

  // ---- pack weights (K ordered kh,kw,ic; NHWC activations; N padded to 64) ----
  int eICp[4] = {8, 96, 128, 192};      // layer-0 IC padded 3 -> 8
  int eICa[4] = {3, 96, 128, 192};
  int eOC[4]  = {96, 128, 192, 256};
  int esz[5]  = {64, 32, 16, 8, 4};
  int eKact[4], eKpad[4];
  uint16_t* wt_e[4];
  for (int i = 0; i < 4; ++i) {
    eKact[i] = eICp[i] * 9;
    eKpad[i] = (eKact[i] + 31) & ~31;
    wt_e[i] = walloc((size_t)pad64(eOC[i]) * eKpad[i]);
    PackP pp; pp.mode = 0; pp.Kpad = eKpad[i]; pp.Kact = eKact[i]; pp.Nact = eOC[i]; pp.Npad = pad64(eOC[i]);
    pp.ICp = eICp[i]; pp.ICact = eICa[i]; pp.KH = 3; pp.KW = 3;
    run_pack(stream, ew[i], wt_e[i], pp);
  }
  uint16_t* wt_fc1 = walloc((size_t)512 * 4096);
  { PackP pp; pp.mode = 3; pp.Kpad = 4096; pp.Kact = 4096; pp.Nact = 512; pp.Npad = 512;
    pp.ICp = 256; pp.ICact = 256; pp.KH = 16; pp.KW = 1;
    run_pack(stream, efw, wt_fc1, pp); }
  uint16_t* wt_fc2 = walloc((size_t)4096 * 512);
  { PackP pp; pp.mode = 4; pp.Kpad = 512; pp.Kact = 512; pp.Nact = 4096; pp.Npad = 4096;
    pp.ICp = 256; pp.ICact = 256; pp.KH = 16; pp.KW = 1;
    run_pack(stream, dfw, wt_fc2, pp); }
  pack_bias_fc2_kernel<<<16, 256, 0, stream>>>(dfb, bias_fc2);
  int dC[6]  = {256, 192, 128, 96, 64, 48};
  int dsz[6] = {4, 8, 16, 32, 64, 128};
  uint16_t* wt_d[5];
  for (int i = 0; i < 5; ++i) {
    int Kact = dC[i] * 16, N = dC[i + 1];
    wt_d[i] = walloc((size_t)pad64(N) * Kact);
    PackP pp; pp.mode = 1; pp.Kpad = Kact; pp.Kact = Kact; pp.Nact = N; pp.Npad = pad64(N);
    pp.ICp = dC[i]; pp.ICact = dC[i]; pp.KH = 4; pp.KW = 4;
    run_pack(stream, dw[i], wt_d[i], pp);
  }
  uint16_t* wt_hx = walloc((size_t)64 * 64);
  { PackP pp; pp.mode = 0; pp.Kpad = 64; pp.Kact = 48; pp.Nact = 24; pp.Npad = 64;
    pp.ICp = 48; pp.ICact = 48; pp.KH = 1; pp.KW = 1;
    run_pack(stream, xyzw, wt_hx, pp); }
  uint16_t* wt_hm = walloc((size_t)64 * 64);
  { PackP pp; pp.mode = 0; pp.Kpad = 64; pp.Kact = 48; pp.Nact = 8; pp.Npad = 64;
    pp.ICp = 48; pp.ICact = 48; pp.KH = 1; pp.KW = 1;
    run_pack(stream, maskw, wt_hm, pp); }

  // ---- input prep: NCHW f32 -> NHWC bf16 padded ----
  { int total = Bn * 64 * 64 * 8;
    prep_x_kernel<<<(total + 255) / 256, 256, 0, stream>>>(x, bufs[1], Bn); }

  const uint16_t* cur = bufs[1];
  int pb = 0;

  // ---- encoder: 4x (conv 3x3 s2, relu) ----
  for (int i = 0; i < 4; ++i) {
    GemmP g = mk_gemm(eICp[i], esz[i], esz[i], eOC[i], esz[i + 1], esz[i + 1],
                      3, 2, 0, eKpad[i], eKact[i], Bn * esz[i + 1] * esz[i + 1], 1, 0);
    run_gemm(stream, cur, wt_e[i], eb[i], bufs[pb], zpad, g, pad64(eOC[i]), 0);
    cur = bufs[pb]; pb ^= 1;
  }
  // ---- fc1: (10,4096 NHWC-flat) -> (10,512), relu ----
  { GemmP g = mk_gemm(4096, 1, 1, 512, 1, 1, 1, 1, 0, 4096, 4096, Bn, 1, 0);
    run_gemm(stream, cur, wt_fc1, efb, bufs[pb], zpad, g, 512, 0);
    cur = bufs[pb]; pb ^= 1; }
  // ---- fc2: (10,512) -> (10,4,4,256) NHWC (N-permuted weights), relu ----
  { GemmP g = mk_gemm(512, 1, 1, 4096, 1, 1, 1, 1, 0, 512, 512, Bn, 1, 0);
    run_gemm(stream, cur, wt_fc2, bias_fc2, bufs[pb], zpad, g, 4096, 0);
    cur = bufs[pb]; pb ^= 1; }
  // ---- decoder: 5x (deconv 4x4 s2, relu) ----
  for (int i = 0; i < 5; ++i) {
    GemmP g = mk_gemm(dC[i], dsz[i], dsz[i], dC[i + 1], dsz[i + 1], dsz[i + 1],
                      4, 2, 2, dC[i] * 16, dC[i] * 16, Bn * dsz[i + 1] * dsz[i + 1], 1, 0);
    run_gemm(stream, cur, wt_d[i], db[i], bufs[pb], zpad, g, pad64(dC[i + 1]), 1);
    cur = bufs[pb]; pb ^= 1;
  }
  // ---- XYZ head: 48 -> 24, bf16 NHWC (10,128,128,24) ----
  uint16_t* xyzbuf = bufs[pb];
  { GemmP g = mk_gemm(48, 128, 128, 24, 128, 128, 1, 1, 0, 64, 48, Bn * 128 * 128, 0, 0);
    run_gemm(stream, cur, wt_hx, xyzb, xyzbuf, zpad, g, 64, 0); }
  // ---- mask head: 48 -> 8, f32 channel-major straight into d_out tail ----
  { GemmP g = mk_gemm(48, 128, 128, 8, 128, 128, 1, 1, 0, 64, 48, Bn * 128 * 128, 0, 1);
    run_gemm(stream, cur, wt_hm, maskb, out + (size_t)Bn * 3 * 8 * 128 * 128, zpad, g, 64, 0); }
  // ---- geometry: K^-1 + R^T ----
  { int total = Bn * 8 * 128 * 128;
    geom_kernel<<<(total + 255) / 256, 256, 0, stream>>>(xyzbuf, quat, out, Bn); }
}
